// EdgeUpdater_69028714381392
// MI455X (gfx1250) — matrix operand order repaired and checked by reading, not compiled, not run on hardware
//
#include <hip/hip_runtime.h>

// EdgeUpdater for MI455X (gfx1250, wave32, WMMA).
// All GEMMs use V_WMMA_F32_16X16X32_F16 with an f16 hi/lo split of every f32
// operand:  x = hi + lo,  A*B ~= Ahi*Bhi + Ahi*Blo + Alo*Bhi  (error ~2^-22).
// 3 wmma per 32-K chunk vs 8 f32 wmma -> 2.7x fewer matrix-issue slots.
// Pipeline:
//   K1: h = x @ W_lin^T + b_lin
//   K2: stats pass: u1 = [h[s]+h[d], ea] @ W1^T + b1 -> column sum/sumsq partials
//   KS: reduce partials -> BN1 scale/shift
//   K3: recompute u1, BN1+ReLU in regs, LDS bounce, u2 = z1 @ W2^T + b2 -> d_out,
//       + partials of u2
//   KS: reduce partials -> BN2 scale/shift
//   K5: d_out = relu(d_out*scale2 + shift2) in place
// Workspace: h (25.6MB) + partials (1.25MB) + 512 floats ~= 27MB.

#define NN   50000
#define EE   640000
#define NIN  128
#define NOUT 128
#define EPSV 1e-5f
#define BLK2 1250  // 1250 blocks * 8 waves * 4 tiles * 16 edges = 640000

typedef _Float16 f16;
typedef __attribute__((ext_vector_type(16))) _Float16 v16h;
typedef __attribute__((ext_vector_type(8)))  _Float16 v8h;
typedef __attribute__((ext_vector_type(4)))  _Float16 v4h;
typedef __attribute__((ext_vector_type(8)))  float    v8f;

// Padded LDS row strides (f16 elems / f32 elems); stride*elemsize/4 mod 64 == 4
// -> B-fragment ds_load_b128 reaches the 2-cycle/512B bank minimum.
#define SW_K128 136   // f16 rows of length 128 (+8 pad)
#define SW_K256 264   // f16 rows of length 256 (+8 pad)
#define SZ_K128 132   // f32 rows of length 128 (+4 pad)

__device__ __forceinline__ v8f wmma16(v16h a, v16h b, v8f c) {
  return __builtin_amdgcn_wmma_f32_16x16x32_f16(false, a, false, b, (short)0, c,
                                                false, false);
}

union F4u { float4 v; float f[4]; };

// Load the two 8-float runs of a 32-K chunk for this lane-half.
__device__ __forceinline__ void load2runs(const float* __restrict__ p, int kc,
                                          float* __restrict__ buf) {
  F4u r0, r1, r2, r3;
  r0.v = *(const float4*)(p + kc);
  r1.v = *(const float4*)(p + kc + 4);
  r2.v = *(const float4*)(p + kc + 16);
  r3.v = *(const float4*)(p + kc + 20);
#pragma unroll
  for (int i = 0; i < 4; ++i) {
    buf[i] = r0.f[i];
    buf[4 + i] = r1.f[i];
    buf[8 + i] = r2.f[i];
    buf[12 + i] = r3.f[i];
  }
}

__device__ __forceinline__ void load2runs_add(const float* __restrict__ p,
                                              const float* __restrict__ q,
                                              int kc, float* __restrict__ buf) {
  F4u p0, p1, p2, p3, q0, q1, q2, q3;
  p0.v = *(const float4*)(p + kc);
  p1.v = *(const float4*)(p + kc + 4);
  p2.v = *(const float4*)(p + kc + 16);
  p3.v = *(const float4*)(p + kc + 20);
  q0.v = *(const float4*)(q + kc);
  q1.v = *(const float4*)(q + kc + 4);
  q2.v = *(const float4*)(q + kc + 16);
  q3.v = *(const float4*)(q + kc + 20);
#pragma unroll
  for (int i = 0; i < 4; ++i) {
    buf[i] = p0.f[i] + q0.f[i];
    buf[4 + i] = p1.f[i] + q1.f[i];
    buf[8 + i] = p2.f[i] + q2.f[i];
    buf[12 + i] = p3.f[i] + q3.f[i];
  }
}

__device__ __forceinline__ void split16(const float* __restrict__ buf, v16h& hi,
                                        v16h& lo) {
#pragma unroll
  for (int j = 0; j < 16; ++j) {
    const float x = buf[j];
    const f16 hh = (f16)x;
    hi[j] = hh;
    lo[j] = (f16)(x - (float)hh);
  }
}

__device__ __forceinline__ v16h ldsB16(const f16* __restrict__ p) {
  const v8h b0 = *(const v8h*)(p);
  const v8h b1 = *(const v8h*)(p + 16);
  return __builtin_shufflevector(b0, b1, 0, 1, 2, 3, 4, 5, 6, 7, 8, 9, 10, 11,
                                 12, 13, 14, 15);
}

// Split one float4 of a weight row into f16 hi/lo and store to padded LDS rows.
__device__ __forceinline__ void stage_split4(f16* __restrict__ dhi,
                                             f16* __restrict__ dlo,
                                             float4 w) {
  F4u u;
  u.v = w;
  v4h hi, lo;
#pragma unroll
  for (int i = 0; i < 4; ++i) {
    const f16 hh = (f16)u.f[i];
    hi[i] = hh;
    lo[i] = (f16)(u.f[i] - (float)hh);
  }
  *(v4h*)dhi = hi;
  *(v4h*)dlo = lo;
}

// ---------------- K1: node linear -------------------------------------------
__global__ __launch_bounds__(256) void k_node_linear(
    const float* __restrict__ x, const float* __restrict__ Wl,
    const float* __restrict__ bl, float* __restrict__ h) {
  extern __shared__ char smemraw[];
  f16* sWhi = (f16*)smemraw;            // 128*SW_K128
  f16* sWlo = sWhi + 128 * SW_K128;
  const int t = threadIdx.x;
  for (int i4 = t; i4 < (NIN * NIN) / 4; i4 += 256) {
    const int n = i4 >> 5;
    const int k = (i4 & 31) << 2;
    stage_split4(sWhi + n * SW_K128 + k, sWlo + n * SW_K128 + k,
                 ((const float4*)Wl)[i4]);
  }
  __syncthreads();
  const int l16 = t & 15, half = (t >> 4) & 1, wave = t >> 5;
  const int row0 = (blockIdx.x * 8 + wave) * 16;
  if (row0 >= NN) return;

  v8f acc[8];
#pragma unroll
  for (int nt = 0; nt < 8; ++nt) {
    const float bv = bl[nt * 16 + l16];
#pragma unroll
    for (int v = 0; v < 8; ++v) acc[nt][v] = bv;
  }
  const float* xrow = x + (size_t)(row0 + l16) * NIN + 8 * half;
#pragma unroll
  for (int c = 0; c < 4; ++c) {  // K = 128 -> 4 chunks of 32
    float buf[16];
    load2runs(xrow, 32 * c, buf);
    v16h ahi, alo;
    split16(buf, ahi, alo);
    const int ko = 8 * half + 32 * c;
#pragma unroll
    for (int nt = 0; nt < 8; ++nt) {
      const v16h bhi = ldsB16(sWhi + (nt * 16 + l16) * SW_K128 + ko);
      const v16h blo = ldsB16(sWlo + (nt * 16 + l16) * SW_K128 + ko);
      acc[nt] = wmma16(ahi, bhi, acc[nt]);
      acc[nt] = wmma16(ahi, blo, acc[nt]);
      acc[nt] = wmma16(alo, bhi, acc[nt]);
    }
  }
#pragma unroll
  for (int nt = 0; nt < 8; ++nt)
#pragma unroll
    for (int v = 0; v < 8; ++v)
      h[(size_t)(row0 + v + 8 * half) * NIN + nt * 16 + l16] = acc[nt][v];
}

// ---------------- K2: MLP layer-1 stats pass --------------------------------
__global__ __launch_bounds__(256) void k_edge_mlp1_stats(
    const float* __restrict__ h, const long long* __restrict__ ei,
    const float* __restrict__ ea, const float* __restrict__ W1,
    const float* __restrict__ b1, float* __restrict__ partials) {
  extern __shared__ char smemraw[];
  f16* sW1hi = (f16*)smemraw;                 // 128*SW_K256
  f16* sW1lo = sW1hi + 128 * SW_K256;
  float* red = (float*)(sW1lo + 128 * SW_K256);  // 8*256
  const int t = threadIdx.x;
  for (int i4 = t; i4 < (128 * 256) / 4; i4 += 256) {
    const int n = i4 >> 6;
    const int k = (i4 & 63) << 2;
    stage_split4(sW1hi + n * SW_K256 + k, sW1lo + n * SW_K256 + k,
                 ((const float4*)W1)[i4]);
  }
  __syncthreads();
  const int l16 = t & 15, half = (t >> 4) & 1, wave = t >> 5;
  float bias[8], s[8], q[8];
#pragma unroll
  for (int nt = 0; nt < 8; ++nt) {
    bias[nt] = b1[nt * 16 + l16];
    s[nt] = 0.f;
    q[nt] = 0.f;
  }
  for (int tt = 0; tt < 4; ++tt) {
    const int e = ((blockIdx.x * 8 + wave) * 4 + tt) * 16 + l16;
    const int src = (int)ei[e];
    const int dst = (int)ei[EE + e];
    const float* hs = h + (size_t)src * NIN + 8 * half;
    const float* hd = h + (size_t)dst * NIN + 8 * half;
    const float* ae = ea + (size_t)e * NIN + 8 * half;
    v8f acc[8];
#pragma unroll
    for (int nt = 0; nt < 8; ++nt)
#pragma unroll
      for (int v = 0; v < 8; ++v) acc[nt][v] = bias[nt];
#pragma unroll
    for (int c = 0; c < 8; ++c) {  // K = 256 -> 8 chunks of 32
      float buf[16];
      if (c < 4)
        load2runs_add(hs, hd, 32 * c, buf);     // gather-add half of z
      else
        load2runs(ae, 32 * (c - 4), buf);       // edge_attr half of z
      v16h ahi, alo;
      split16(buf, ahi, alo);
      const int ko = 8 * half + 32 * c;
#pragma unroll
      for (int nt = 0; nt < 8; ++nt) {
        const v16h bhi = ldsB16(sW1hi + (nt * 16 + l16) * SW_K256 + ko);
        const v16h blo = ldsB16(sW1lo + (nt * 16 + l16) * SW_K256 + ko);
        acc[nt] = wmma16(ahi, bhi, acc[nt]);
        acc[nt] = wmma16(ahi, blo, acc[nt]);
        acc[nt] = wmma16(alo, bhi, acc[nt]);
      }
    }
#pragma unroll
    for (int nt = 0; nt < 8; ++nt) {
      float ls = 0.f, lq = 0.f;
#pragma unroll
      for (int v = 0; v < 8; ++v) {
        const float u = acc[nt][v];
        ls += u;
        lq += u * u;
      }
      s[nt] += ls;
      q[nt] += lq;
    }
  }
#pragma unroll
  for (int nt = 0; nt < 8; ++nt) {
    s[nt] += __shfl_xor(s[nt], 16, 32);
    q[nt] += __shfl_xor(q[nt], 16, 32);
    if (half == 0) {
      red[wave * 256 + nt * 16 + l16] = s[nt];
      red[wave * 256 + 128 + nt * 16 + l16] = q[nt];
    }
  }
  __syncthreads();
  float tot = 0.f;
#pragma unroll
  for (int w = 0; w < 8; ++w) tot += red[w * 256 + t];
  partials[(size_t)blockIdx.x * 256 + t] = tot;
}

// ---------------- KS: reduce partials -> BN scale/shift ---------------------
__global__ void k_bn_stats(const float* __restrict__ partials, int P,
                           const float* __restrict__ g,
                           const float* __restrict__ be,
                           float* __restrict__ scale, float* __restrict__ shift) {
  const int c = threadIdx.x;  // 0..127
  float s = 0.f, q = 0.f;
  for (int p = 0; p < P; ++p) {
    s += partials[p * 256 + c];
    q += partials[p * 256 + 128 + c];
  }
  const float inv = 1.0f / (float)EE;
  const float mu = s * inv;
  const float var = fmaxf(q * inv - mu * mu, 0.f);
  const float sc = g[c] * rsqrtf(var + EPSV);
  scale[c] = sc;
  shift[c] = be[c] - mu * sc;
}

// ---------------- K3: fused recompute + BN1 + ReLU + layer-2 ----------------
__global__ __launch_bounds__(256) void k_edge_mlp_fused(
    const float* __restrict__ h, const long long* __restrict__ ei,
    const float* __restrict__ ea, const float* __restrict__ W1,
    const float* __restrict__ b1, const float* __restrict__ sc1,
    const float* __restrict__ sh1, const float* __restrict__ W2,
    const float* __restrict__ b2, float* __restrict__ out,
    float* __restrict__ partials) {
  extern __shared__ char smemraw[];
  f16* sW1hi = (f16*)smemraw;                       // 128*SW_K256
  f16* sW1lo = sW1hi + 128 * SW_K256;
  f16* sW2hi = sW1lo + 128 * SW_K256;               // 128*SW_K128
  f16* sW2lo = sW2hi + 128 * SW_K128;
  float* sZ = (float*)(sW2lo + 128 * SW_K128);      // 8 waves * 16*SZ_K128
  float* red = sZ + 8 * 16 * SZ_K128;               // 8*256
  const int t = threadIdx.x;
  for (int i4 = t; i4 < (128 * 256) / 4; i4 += 256) {
    const int n = i4 >> 6;
    const int k = (i4 & 63) << 2;
    stage_split4(sW1hi + n * SW_K256 + k, sW1lo + n * SW_K256 + k,
                 ((const float4*)W1)[i4]);
  }
  for (int i4 = t; i4 < (128 * 128) / 4; i4 += 256) {
    const int n = i4 >> 5;
    const int k = (i4 & 31) << 2;
    stage_split4(sW2hi + n * SW_K128 + k, sW2lo + n * SW_K128 + k,
                 ((const float4*)W2)[i4]);
  }
  __syncthreads();
  const int l16 = t & 15, half = (t >> 4) & 1, wave = t >> 5;
  float* zw = sZ + wave * 16 * SZ_K128;  // per-wave private D->A bounce
  float bias1[8], scale1[8], shift1[8], bias2[8], s[8], q[8];
#pragma unroll
  for (int nt = 0; nt < 8; ++nt) {
    bias1[nt] = b1[nt * 16 + l16];
    scale1[nt] = sc1[nt * 16 + l16];
    shift1[nt] = sh1[nt * 16 + l16];
    bias2[nt] = b2[nt * 16 + l16];
    s[nt] = 0.f;
    q[nt] = 0.f;
  }
  for (int tt = 0; tt < 4; ++tt) {
    const int e0 = ((blockIdx.x * 8 + wave) * 4 + tt) * 16;
    const int e = e0 + l16;
    const int src = (int)ei[e];
    const int dst = (int)ei[EE + e];
    const float* hs = h + (size_t)src * NIN + 8 * half;
    const float* hd = h + (size_t)dst * NIN + 8 * half;
    const float* ae = ea + (size_t)e * NIN + 8 * half;
    v8f acc[8];
#pragma unroll
    for (int nt = 0; nt < 8; ++nt)
#pragma unroll
      for (int v = 0; v < 8; ++v) acc[nt][v] = bias1[nt];
#pragma unroll
    for (int c = 0; c < 8; ++c) {
      float buf[16];
      if (c < 4)
        load2runs_add(hs, hd, 32 * c, buf);
      else
        load2runs(ae, 32 * (c - 4), buf);
      v16h ahi, alo;
      split16(buf, ahi, alo);
      const int ko = 8 * half + 32 * c;
#pragma unroll
      for (int nt = 0; nt < 8; ++nt) {
        const v16h bhi = ldsB16(sW1hi + (nt * 16 + l16) * SW_K256 + ko);
        const v16h blo = ldsB16(sW1lo + (nt * 16 + l16) * SW_K256 + ko);
        acc[nt] = wmma16(ahi, bhi, acc[nt]);
        acc[nt] = wmma16(ahi, blo, acc[nt]);
        acc[nt] = wmma16(alo, bhi, acc[nt]);
      }
    }
    // BN1 + ReLU in D-layout, bounce through LDS into A-layout for GEMM2.
#pragma unroll
    for (int nt = 0; nt < 8; ++nt)
#pragma unroll
      for (int v = 0; v < 8; ++v) {
        const float z = fmaxf(acc[nt][v] * scale1[nt] + shift1[nt], 0.f);
        zw[(v + 8 * half) * SZ_K128 + nt * 16 + l16] = z;
      }
    __syncthreads();  // uniform across all 8 waves every iteration
    v8f acc2[8];
#pragma unroll
    for (int nt = 0; nt < 8; ++nt)
#pragma unroll
      for (int v = 0; v < 8; ++v) acc2[nt][v] = bias2[nt];
    const float* zrow = zw + l16 * SZ_K128 + 8 * half;
#pragma unroll
    for (int c = 0; c < 4; ++c) {
      float buf[16];
      load2runs(zrow, 32 * c, buf);
      v16h ahi, alo;
      split16(buf, ahi, alo);
      const int ko = 8 * half + 32 * c;
#pragma unroll
      for (int nt = 0; nt < 8; ++nt) {
        const v16h bhi = ldsB16(sW2hi + (nt * 16 + l16) * SW_K128 + ko);
        const v16h blo = ldsB16(sW2lo + (nt * 16 + l16) * SW_K128 + ko);
        acc2[nt] = wmma16(ahi, bhi, acc2[nt]);
        acc2[nt] = wmma16(ahi, blo, acc2[nt]);
        acc2[nt] = wmma16(alo, bhi, acc2[nt]);
      }
    }
#pragma unroll
    for (int nt = 0; nt < 8; ++nt) {
      float ls = 0.f, lq = 0.f;
#pragma unroll
      for (int v = 0; v < 8; ++v) {
        const float u = acc2[nt][v];
        out[(size_t)(e0 + v + 8 * half) * NOUT + nt * 16 + l16] = u;
        ls += u;
        lq += u * u;
      }
      s[nt] += ls;
      q[nt] += lq;
    }
  }
#pragma unroll
  for (int nt = 0; nt < 8; ++nt) {
    s[nt] += __shfl_xor(s[nt], 16, 32);
    q[nt] += __shfl_xor(q[nt], 16, 32);
    if (half == 0) {
      red[wave * 256 + nt * 16 + l16] = s[nt];
      red[wave * 256 + 128 + nt * 16 + l16] = q[nt];
    }
  }
  __syncthreads();
  float tot = 0.f;
#pragma unroll
  for (int w = 0; w < 8; ++w) tot += red[w * 256 + t];
  partials[(size_t)blockIdx.x * 256 + t] = tot;
}

// ---------------- K5: in-place BN2 + ReLU -----------------------------------
__global__ __launch_bounds__(256) void k_bn2_relu(float* __restrict__ out,
                                                  const float* __restrict__ scale,
                                                  const float* __restrict__ shift) {
  const size_t i4 = (size_t)blockIdx.x * 256 + threadIdx.x;
  const int c = (int)((i4 << 2) & (NOUT - 1));
  float4 v = ((float4*)out)[i4];
  const float4 sc = *(const float4*)(scale + c);
  const float4 sh = *(const float4*)(shift + c);
  v.x = fmaxf(v.x * sc.x + sh.x, 0.f);
  v.y = fmaxf(v.y * sc.y + sh.y, 0.f);
  v.z = fmaxf(v.z * sc.z + sh.z, 0.f);
  v.w = fmaxf(v.w * sc.w + sh.w, 0.f);
  ((float4*)out)[i4] = v;
}

extern "C" void kernel_launch(void* const* d_in, const int* in_sizes, int n_in,
                              void* d_out, int out_size, void* d_ws, size_t ws_size,
                              hipStream_t stream) {
  (void)in_sizes; (void)n_in; (void)out_size; (void)ws_size;
  const float* x = (const float*)d_in[0];
  const long long* ei = (const long long*)d_in[1];
  const float* ea = (const float*)d_in[2];
  const float* Wl = (const float*)d_in[3];
  const float* bl = (const float*)d_in[4];
  const float* W1 = (const float*)d_in[5];
  const float* b1 = (const float*)d_in[6];
  const float* g1 = (const float*)d_in[7];
  const float* be1 = (const float*)d_in[8];
  const float* W2 = (const float*)d_in[9];
  const float* b2 = (const float*)d_in[10];
  const float* g2 = (const float*)d_in[11];
  const float* be2 = (const float*)d_in[12];
  float* out = (float*)d_out;

  float* wsf = (float*)d_ws;
  float* h = wsf;                              // 50000*128
  float* partials = wsf + (size_t)NN * NIN;    // 1250*256
  float* sc1 = partials + (size_t)BLK2 * 256;  // 128
  float* sh1 = sc1 + 128;
  float* sc2 = sh1 + 128;
  float* sh2 = sc2 + 128;

  const size_t smem1 = (size_t)(2 * 128 * SW_K128) * sizeof(f16);
  k_node_linear<<<dim3((NN + 127) / 128), dim3(256), smem1, stream>>>(x, Wl, bl, h);

  const size_t smem2 =
      (size_t)(2 * 128 * SW_K256) * sizeof(f16) + (size_t)(8 * 256) * sizeof(float);
  k_edge_mlp1_stats<<<dim3(BLK2), dim3(256), smem2, stream>>>(h, ei, ea, W1, b1,
                                                              partials);
  k_bn_stats<<<dim3(1), dim3(128), 0, stream>>>(partials, BLK2, g1, be1, sc1, sh1);

  const size_t smem3 = (size_t)(2 * 128 * SW_K256 + 2 * 128 * SW_K128) * sizeof(f16) +
                       (size_t)(8 * 16 * SZ_K128 + 8 * 256) * sizeof(float);
  k_edge_mlp_fused<<<dim3(BLK2), dim3(256), smem3, stream>>>(
      h, ei, ea, W1, b1, sc1, sh1, W2, b2, out, partials);
  k_bn_stats<<<dim3(1), dim3(128), 0, stream>>>(partials, BLK2, g2, be2, sc2, sh2);

  k_bn2_relu<<<dim3((unsigned)((size_t)EE * NOUT / 1024)), dim3(256), 0, stream>>>(
      out, sc2, sh2);
}
